// LSTM_12652973654290
// MI455X (gfx1250) — matrix operand order repaired and checked by reading, not compiled, not run on hardware
//
#include <hip/hip_runtime.h>
#include <hip/hip_bf16.h>

typedef __attribute__((ext_vector_type(16))) __bf16 v16bf;
typedef __attribute__((ext_vector_type(8)))  float  v8f;

#define kB   256
#define kH   1024
#define kV   32000
#define kT   8
#define kG4H 4096

__device__ __forceinline__ float sigmoidf_(float x) { return 1.0f / (1.0f + __expf(-x)); }

// Round-to-nearest-even float -> bf16 bits
__device__ __forceinline__ unsigned short f2bf_bits(float f) {
  unsigned int u = __float_as_uint(f);
  unsigned int r = u + 0x7FFFu + ((u >> 16) & 1u);
  return (unsigned short)(r >> 16);
}

// Raw fp32 A-tile fetch (two 32B chunks per lane) for one wave's 16x32 tile.
//   lanes 0-15 and 16-31 both hold M = lane&15
//   element i -> K = k0 + 8*(lane>>4) + i          for i in [0,8)
//   element i -> K = k0 + 16 + 8*(lane>>4) + (i-8) for i in [8,16)
__device__ __forceinline__ void fetch_a_f32(const float* __restrict__ W,
                                            int row, int k0, int g, float4 fa[4]) {
  const float* p = W + (size_t)row * kH + k0 + 8 * g;
  fa[0] = *(const float4*)(p + 0);
  fa[1] = *(const float4*)(p + 4);
  fa[2] = *(const float4*)(p + 16);
  fa[3] = *(const float4*)(p + 20);
}

__device__ __forceinline__ v16bf cvt_a_bf16(const float4 fa[4]) {
  v16bf a;
  a[0]  = (__bf16)fa[0].x; a[1]  = (__bf16)fa[0].y; a[2]  = (__bf16)fa[0].z; a[3]  = (__bf16)fa[0].w;
  a[4]  = (__bf16)fa[1].x; a[5]  = (__bf16)fa[1].y; a[6]  = (__bf16)fa[1].z; a[7]  = (__bf16)fa[1].w;
  a[8]  = (__bf16)fa[2].x; a[9]  = (__bf16)fa[2].y; a[10] = (__bf16)fa[2].z; a[11] = (__bf16)fa[2].w;
  a[12] = (__bf16)fa[3].x; a[13] = (__bf16)fa[3].y; a[14] = (__bf16)fa[3].z; a[15] = (__bf16)fa[3].w;
  return a;
}

// Shared software-pipelined GEMM core:
//   acc[bt] (+)= A(16 rows of W, 32 k) x B(hT panel in LDS, 32 k x 256 b)
// for all 32 k-panels. lds must be 32-aligned, 16 KB.
__device__ __forceinline__ void gemm_core(const float* __restrict__ W, int row0, int g,
                                          const unsigned short* __restrict__ hT,
                                          unsigned short* lds, int tid, int lane,
                                          v8f acc[16]) {
  const uint4* src = (const uint4*)hT;   // 1024 uint4 per 32-row panel
  uint4* dst = (uint4*)lds;

  uint4  hstage[4];
  float4 fa[4];
#pragma unroll
  for (int r = 0; r < 4; ++r) hstage[r] = src[tid + 256 * r];
  fetch_a_f32(W, row0, 0, g, fa);

  for (int k0 = 0; k0 < kH; k0 += 32) {
    __syncthreads();   // all waves done reading previous LDS panel
#pragma unroll
    for (int r = 0; r < 4; ++r) dst[tid + 256 * r] = hstage[r];
    __syncthreads();   // panel visible to all waves

    v16bf a = cvt_a_bf16(fa);
    if (k0 + 32 < kH) {
      const uint4* nsrc = src + (k0 + 32) * 32;
#pragma unroll
      for (int r = 0; r < 4; ++r) hstage[r] = nsrc[tid + 256 * r];
      fetch_a_f32(W, row0, k0 + 32, g, fa);
    }
#pragma unroll
    for (int bt = 0; bt < 16; ++bt) {
      v16bf bm;
#pragma unroll
      for (int s = 0; s < 16; ++s)
        bm[s] = __builtin_bit_cast(__bf16, lds[(8 * g + (s & 7) + 16 * (s >> 3)) * kB + bt * 16 + (lane & 15)]);
      acc[bt] = __builtin_amdgcn_wmma_f32_16x16x32_bf16(
          false, a, false, bm, (short)0, acc[bt], false, false);
    }
  }
}

// pre[j][b] = sum_k hT_prev[k][b] * Wh_t[j][k]  (+ x[b,t]*Wx_t[j] + bh_t[j])
// One block = 8 waves = 128 j rows, all 256 b columns. 32 blocks cover j=4096.
__global__ void __launch_bounds__(256)
lstm_pre_gemm(const float* __restrict__ Wh_t,             // (4096, 1024)
              const float* __restrict__ Wx_t,             // (4096,)
              const float* __restrict__ bh_t,             // (4096,)
              const float* __restrict__ x,                // (B, 8)
              const unsigned short* __restrict__ hT_prev, // (1024, 256) bf16
              float* __restrict__ preT,                   // (4096, 256)
              int t, int has_prev) {
  __shared__ __align__(32) unsigned short lds_h[32 * kB];   // 16 KB
  const int tid  = threadIdx.x;
  const int lane = tid & 31;
  const int wave = tid >> 5;
  const int g    = lane >> 4;
  const int l15  = lane & 15;
  const int j0   = blockIdx.x * 128 + wave * 16;

  v8f zero = {};
  v8f acc[16];
#pragma unroll
  for (int i = 0; i < 16; ++i) acc[i] = zero;

  if (has_prev)
    gemm_core(Wh_t, j0 + l15, g, hT_prev, lds_h, tid, lane, acc);

  // Epilogue: pre += x_t * Wx + bh  (bh only when a previous h exists)
  const int jw = j0 + 8 * g;
  float wx[8], bb[8];
#pragma unroll
  for (int r = 0; r < 8; ++r) {
    wx[r] = Wx_t[jw + r];
    bb[r] = has_prev ? bh_t[jw + r] : 0.0f;
  }
#pragma unroll
  for (int bt = 0; bt < 16; ++bt) {
    int b = bt * 16 + l15;
    float xv = x[b * kT + t];
#pragma unroll
    for (int r = 0; r < 8; ++r)
      preT[(size_t)(jw + r) * kB + b] = acc[bt][r] + xv * wx[r] + bb[r];
  }
}

// Gate nonlinearity + cell/hidden update. Writes transposed bf16 h for GEMMs.
__global__ void __launch_bounds__(256)
lstm_gates(const float* __restrict__ preT,    // (4096, 256)
           float* __restrict__ cT,            // (1024, 256)
           unsigned short* __restrict__ hT_t, // (1024, 256) bf16
           int has_prev) {
  int idx = blockIdx.x * 256 + threadIdx.x;
  int b = idx & (kB - 1);
  int u = idx >> 8;
  float pf = preT[(size_t)(0 * kH + u) * kB + b];
  float pi = preT[(size_t)(1 * kH + u) * kB + b];
  float pc = preT[(size_t)(2 * kH + u) * kB + b];
  float po = preT[(size_t)(3 * kH + u) * kB + b];
  float f = sigmoidf_(pf);
  float i = sigmoidf_(pi);
  float cb = tanhf(pc);
  float o = sigmoidf_(po);
  float cn = has_prev ? (f * cT[(size_t)u * kB + b] + i * cb) : (i * cb);
  cT[(size_t)u * kB + b] = cn;
  float h = o * tanhf(cn);
  hT_t[(size_t)u * kB + b] = f2bf_bits(h);
}

// logits[b][t][v] = sum_k h_t[b][k] * Wout_t[v][k] + bout_t[v]
// One block = 8 waves = 128 v rows x all 256 b. 250 blocks per t, 8 t.
__global__ void __launch_bounds__(256)
lstm_proj(const float* __restrict__ Wout,             // (8, 32000, 1024)
          const float* __restrict__ bout,             // (8, 32000)
          const unsigned short* __restrict__ hT_base, // 8 x (1024,256) bf16
          float* __restrict__ out) {                  // (B, 8, V)
  __shared__ __align__(32) unsigned short lds_h[32 * kB];
  const int tid  = threadIdx.x;
  const int lane = tid & 31;
  const int wave = tid >> 5;
  const int g    = lane >> 4;
  const int l15  = lane & 15;
  const int t    = blockIdx.x / 250;
  const int vblk = blockIdx.x % 250;
  const int v0   = vblk * 128 + wave * 16;

  const float* Wout_t = Wout + (size_t)t * kV * kH;
  const unsigned short* hT_t = hT_base + (size_t)t * kH * kB;

  v8f zero = {};
  v8f acc[16];
#pragma unroll
  for (int i = 0; i < 16; ++i) acc[i] = zero;

  gemm_core(Wout_t, v0 + l15, g, hT_t, lds_h, tid, lane, acc);

  const int vw = v0 + 8 * g;
  float4 bo0 = *(const float4*)(bout + (size_t)t * kV + vw);
  float4 bo1 = *(const float4*)(bout + (size_t)t * kV + vw + 4);
#pragma unroll
  for (int bt = 0; bt < 16; ++bt) {
    int b = bt * 16 + l15;
    float* dst = out + ((size_t)(b * kT + t)) * kV + vw;
    float4 lo = make_float4(acc[bt][0] + bo0.x, acc[bt][1] + bo0.y,
                            acc[bt][2] + bo0.z, acc[bt][3] + bo0.w);
    float4 hi = make_float4(acc[bt][4] + bo1.x, acc[bt][5] + bo1.y,
                            acc[bt][6] + bo1.z, acc[bt][7] + bo1.w);
    *(float4*)(dst) = lo;
    *(float4*)(dst + 4) = hi;
  }
}

extern "C" void kernel_launch(void* const* d_in, const int* in_sizes, int n_in,
                              void* d_out, int out_size, void* d_ws, size_t ws_size,
                              hipStream_t stream) {
  (void)in_sizes; (void)n_in; (void)out_size; (void)ws_size;
  const float* x    = (const float*)d_in[0];   // (256, 8)
  const float* Wx   = (const float*)d_in[1];   // (8, 4, 1024, 1)
  const float* Wh   = (const float*)d_in[2];   // (8, 4, 1024, 1024)
  const float* bh   = (const float*)d_in[3];   // (8, 4, 1024)
  const float* Wout = (const float*)d_in[4];   // (8, 32000, 1024)
  const float* bout = (const float*)d_in[5];   // (8, 32000)
  float* out = (float*)d_out;                  // (256, 8, 32000)

  char* ws = (char*)d_ws;
  float*          preT = (float*)(ws);                       // 4 MiB
  float*          cT   = (float*)(ws + (4u << 20));          // 1 MiB
  unsigned short* hT   = (unsigned short*)(ws + (5u << 20)); // 8 * 512 KiB bf16

  for (int t = 0; t < kT; ++t) {
    int has_prev = (t > 0) ? 1 : 0;
    const unsigned short* hT_prev =
        hT + (size_t)(has_prev ? (t - 1) : 0) * kH * kB;
    lstm_pre_gemm<<<32, 256, 0, stream>>>(
        Wh + (size_t)t * kG4H * kH, Wx + (size_t)t * kG4H,
        bh + (size_t)t * kG4H, x, hT_prev, preT, t, has_prev);
    lstm_gates<<<1024, 256, 0, stream>>>(
        preT, cT, hT + (size_t)t * kH * kB, has_prev);
  }
  lstm_proj<<<2000, 256, 0, stream>>>(Wout, bout, hT, out);
}
